// MultiHeadSelfAttention_46986942218860
// MI455X (gfx1250) — matrix operand order repaired and checked by reading, not compiled, not run on hardware
//
#include <hip/hip_runtime.h>
#include <stdint.h>

// ---------------------------------------------------------------------------
// Types for WMMA
// ---------------------------------------------------------------------------
typedef __attribute__((ext_vector_type(16))) __bf16 bf16x16;
typedef __attribute__((ext_vector_type(8)))  float  f32x8;

union FragB {
    bf16x16 v;
    uint4   u[2];   // 32 bytes = 16 bf16
};

__device__ __forceinline__ unsigned short f2bf(float f) {
    union { float f; unsigned u; } x; x.f = f;
    unsigned r = x.u + 0x7FFFu + ((x.u >> 16) & 1u);   // round-to-nearest-even
    return (unsigned short)(r >> 16);
}

#define WMMA_BF16(A, B, C) \
    __builtin_amdgcn_wmma_f32_16x16x32_bf16(false, (A), false, (B), (short)0, (C), false, false)

// ---------------------------------------------------------------------------
// fp32 -> bf16 conversion (vectorized by 4)
// ---------------------------------------------------------------------------
__global__ void cvt_f32_bf16(const float* __restrict__ s,
                             unsigned short* __restrict__ d, int n4) {
    int i = blockIdx.x * blockDim.x + threadIdx.x;
    if (i < n4) {
        float4 f = ((const float4*)s)[i];
        ushort4 o;
        o.x = f2bf(f.x); o.y = f2bf(f.y); o.z = f2bf(f.z); o.w = f2bf(f.w);
        ((ushort4*)d)[i] = o;
    }
}

// ---------------------------------------------------------------------------
// GEMM: C[M,N] = A[M,K] (bf16, row-major) * W[N,K]^T (bf16, row-major) + bias
// Block: 256 threads (8 waves). Block tile 128(M) x 128(N). Wave tile 32x64:
// 8 WMMAs per K-step from 12 ds_load_b128 (1.5 LDS reads per WMMA).
// Staging: 4 named-temp b128 loads so all 4 stay in flight (no serialization).
// ---------------------------------------------------------------------------
#define GLDS 40   // padded LDS row stride (halves): 80B rows -> conflict-free

__global__ __launch_bounds__(256)
void gemm_bf16nt(const unsigned short* __restrict__ A,
                 const unsigned short* __restrict__ W,
                 const float* __restrict__ bias,
                 float* __restrict__ Cf,            // fp32 output (or null)
                 unsigned short* __restrict__ Cb,   // bf16 output (if Cf null)
                 int M, int N, int K) {
    __shared__ __align__(16) unsigned short As[128 * GLDS];
    __shared__ __align__(16) unsigned short Ws[128 * GLDS];

    const int tid  = threadIdx.x;
    const int lane = tid & 31;
    const int wave = tid >> 5;
    const int l16  = lane & 15;
    const int koffA = (lane < 16) ? 0 : 8;    // A-frag K pattern: {0-7,16-23} / {8-15,24-31}
    const int koffB = (lane < 16) ? 0 : 16;   // unused: B-frags load with koffA, same K pattern as A
    const int rbase = (lane < 16) ? 0 : 8;    // C-layout row base
    const int wm = wave >> 1, wn = wave & 1;  // 4 x 2 wave grid

    const long mBase = (long)blockIdx.y * 128;
    const long nBase = (long)blockIdx.x * 128;

    f32x8 acc[2][4];
#pragma unroll
    for (int a = 0; a < 2; ++a)
#pragma unroll
        for (int b = 0; b < 4; ++b)
#pragma unroll
            for (int i = 0; i < 8; ++i) acc[a][b][i] = 0.0f;

    // Per-thread staging coordinates: 512 uint4 per 128x32 slab, 2 per thread.
    const int sr  = tid >> 2, ss = tid & 3;          // chunk 0: row sr, seg ss
    const int sr2 = (tid + 256) >> 2;                // chunk 1

    const unsigned short* gA0 = A + (mBase + sr)  * (long)K + ss * 8;
    const unsigned short* gA1 = A + (mBase + sr2) * (long)K + ss * 8;
    const unsigned short* gW0 = W + (nBase + sr)  * (long)K + ss * 8;
    const unsigned short* gW1 = W + (nBase + sr2) * (long)K + ss * 8;
    unsigned short* sA0 = As + sr  * GLDS + ss * 8;
    unsigned short* sA1 = As + sr2 * GLDS + ss * 8;
    unsigned short* sW0 = Ws + sr  * GLDS + ss * 8;
    unsigned short* sW1 = Ws + sr2 * GLDS + ss * 8;

    for (int kk = 0; kk < K; kk += 32) {
        // Issue all 4 staging loads before any LDS store -> 4 loads in flight.
        uint4 t0 = *(const uint4*)(gA0 + kk);
        uint4 t1 = *(const uint4*)(gA1 + kk);
        uint4 t2 = *(const uint4*)(gW0 + kk);
        uint4 t3 = *(const uint4*)(gW1 + kk);

        // Prefetch next K-slab toward the WGP (global_prefetch_b8).
        if (kk + 32 < K) {
            __builtin_prefetch(gA0 + kk + 32, 0, 3);
            __builtin_prefetch(gW0 + kk + 32, 0, 3);
        }

        *(uint4*)sA0 = t0;
        *(uint4*)sA1 = t1;
        *(uint4*)sW0 = t2;
        *(uint4*)sW1 = t3;
        __syncthreads();

        FragB af[2], bf[4];
#pragma unroll
        for (int mi = 0; mi < 2; ++mi) {
            const unsigned short* p = As + (wm * 32 + mi * 16 + l16) * GLDS + koffA;
            af[mi].u[0] = *(const uint4*)(p);
            af[mi].u[1] = *(const uint4*)(p + 16);
        }
#pragma unroll
        for (int nj = 0; nj < 4; ++nj) {
            const unsigned short* p = Ws + (wn * 64 + nj * 16 + l16) * GLDS + koffA;
            bf[nj].u[0] = *(const uint4*)(p);
            bf[nj].u[1] = *(const uint4*)(p + 16);
        }
#pragma unroll
        for (int mi = 0; mi < 2; ++mi)
#pragma unroll
            for (int nj = 0; nj < 4; ++nj)
                acc[mi][nj] = WMMA_BF16(af[mi].v, bf[nj].v, acc[mi][nj]);
        __syncthreads();
    }

    // Epilogue: bias + store.
#pragma unroll
    for (int mi = 0; mi < 2; ++mi) {
#pragma unroll
        for (int nj = 0; nj < 4; ++nj) {
            long col = nBase + wn * 64 + nj * 16 + l16;
            float bv = bias[col];
#pragma unroll
            for (int i = 0; i < 8; ++i) {
                long row = mBase + wm * 32 + mi * 16 + rbase + i;
                float v = acc[mi][nj][i] + bv;
                if (Cf) Cf[row * (long)N + col] = v;
                else    Cb[row * (long)N + col] = f2bf(v);
            }
        }
    }
}

// ---------------------------------------------------------------------------
// Fused flash-attention with adjacency-scaled scores + masked renorm softmax.
// One 128-thread block (4 waves) per (b, h, 64-row q-block).
// Wave w owns q rows [qb*64 + w*16, +16). LDS: K block, V^T block, P buffer.
// ---------------------------------------------------------------------------
#define ALDS 72   // padded stride: 144B rows -> conflict-free
#define T_ 1024

__global__ __launch_bounds__(128)
void attn_fused(const unsigned short* __restrict__ proj, // (B*T, 3072) bf16: [Q|K|V]
                const float* __restrict__ adj,           // (B, T, T)
                const float* __restrict__ mask,          // (B, T)
                unsigned short* __restrict__ outb)       // (B*T, 1024) bf16
{
    const int b  = blockIdx.z;
    const int h  = blockIdx.y;
    const int qb = blockIdx.x;

    const int tid  = threadIdx.x;
    const int lane = tid & 31;
    const int wave = tid >> 5;
    const int l16  = lane & 15;
    const int koffA = (lane < 16) ? 0 : 8;
    const int koffB = (lane < 16) ? 0 : 16;
    const int rbase = (lane < 16) ? 0 : 8;

    __shared__ __align__(16) unsigned short Kb[64 * ALDS];
    __shared__ __align__(16) unsigned short Vt[64 * ALDS];
    __shared__ __align__(16) unsigned short Pb[64 * ALDS];

    const long rowQ = (long)b * T_ + qb * 64 + wave * 16;  // strip base (global row)

    // Q fragments for this wave's strip (dh = 64 -> two K-steps of 32).
    FragB qf[2];
    {
        const unsigned short* qp = proj + (rowQ + l16) * 3072 + h * 64;
        qf[0].u[0] = *(const uint4*)(qp + koffA);
        qf[0].u[1] = *(const uint4*)(qp + koffA + 16);
        qf[1].u[0] = *(const uint4*)(qp + 32 + koffA);
        qf[1].u[1] = *(const uint4*)(qp + 32 + koffA + 16);
    }

    f32x8 oacc[4];
#pragma unroll
    for (int j = 0; j < 4; ++j)
#pragma unroll
        for (int i = 0; i < 8; ++i) oacc[j][i] = 0.0f;

    float mrun[8], lrun[8];
#pragma unroll
    for (int i = 0; i < 8; ++i) { mrun[i] = -3.0e38f; lrun[i] = 0.0f; }

    const float invS = 0.125f;   // 1/sqrt(64)

    for (int kb = 0; kb < T_ / 64; ++kb) {
        const int k0 = kb * 64;

        // ---- Stage K block (row-major) and V block (transposed) into LDS ----
        {
            int r    = tid & 63;
            int half = tid >> 6;             // d-range 0-31 / 32-63
            const long src = ((long)b * T_ + k0 + r) * 3072 + h * 64 + half * 32;
            const uint4* kp = (const uint4*)(proj + src + 1024);
            const uint4* vp = (const uint4*)(proj + src + 2048);

            // Issue all 8 loads (K + V) before any LDS store.
            uint4 k0v = kp[0], k1v = kp[1], k2v = kp[2], k3v = kp[3];
            uint4 vv[4];
            vv[0] = vp[0]; vv[1] = vp[1]; vv[2] = vp[2]; vv[3] = vp[3];

            uint4* kd = (uint4*)(Kb + r * ALDS + half * 32);
            kd[0] = k0v; kd[1] = k1v; kd[2] = k2v; kd[3] = k3v;

            const unsigned short* vs = (const unsigned short*)vv;
#pragma unroll
            for (int i = 0; i < 32; ++i)
                Vt[(half * 32 + i) * ALDS + r] = vs[i];
        }
        __syncthreads();

        // ---- S = Q @ K^T for this wave's 16x64 strip ----
        f32x8 sacc[4];
#pragma unroll
        for (int j = 0; j < 4; ++j)
#pragma unroll
            for (int i = 0; i < 8; ++i) sacc[j][i] = 0.0f;

#pragma unroll
        for (int ks = 0; ks < 2; ++ks) {
#pragma unroll
            for (int j = 0; j < 4; ++j) {
                FragB bj;
                const unsigned short* p = Kb + (j * 16 + l16) * ALDS + ks * 32 + koffA;
                bj.u[0] = *(const uint4*)(p);
                bj.u[1] = *(const uint4*)(p + 16);
                sacc[j] = WMMA_BF16(qf[ks].v, bj.v, sacc[j]);
            }
        }

        // ---- scale * adj * mask, online softmax ----
        float pv[4][8];
        float rmax[8];
#pragma unroll
        for (int i = 0; i < 8; ++i) rmax[i] = -3.0e38f;

#pragma unroll
        for (int j = 0; j < 4; ++j) {
            int kcol = k0 + j * 16 + l16;
            float mk = mask[b * T_ + kcol];
#pragma unroll
            for (int i = 0; i < 8; ++i) {
                int  qt = qb * 64 + wave * 16 + rbase + i;
                float av = adj[((long)b * T_ + qt) * T_ + kcol];
                float s  = sacc[j][i] * invS * av * mk;
                pv[j][i] = s;
                rmax[i]  = fmaxf(rmax[i], s);
            }
        }
#pragma unroll
        for (int i = 0; i < 8; ++i) {
            float v = rmax[i];
            v = fmaxf(v, __shfl_xor(v, 1, 16));
            v = fmaxf(v, __shfl_xor(v, 2, 16));
            v = fmaxf(v, __shfl_xor(v, 4, 16));
            v = fmaxf(v, __shfl_xor(v, 8, 16));
            rmax[i] = v;
        }
        float scal[8], rsum[8];
#pragma unroll
        for (int i = 0; i < 8; ++i) {
            float nm = fmaxf(mrun[i], rmax[i]);
            scal[i]  = __expf(mrun[i] - nm);
            mrun[i]  = nm;
            rsum[i]  = 0.0f;
        }
#pragma unroll
        for (int j = 0; j < 4; ++j) {
            int kcol = k0 + j * 16 + l16;
            float mk = mask[b * T_ + kcol];
#pragma unroll
            for (int i = 0; i < 8; ++i) {
                float e = __expf(pv[j][i] - mrun[i]) * mk;
                pv[j][i] = e;
                rsum[i] += e;
            }
        }
#pragma unroll
        for (int i = 0; i < 8; ++i) {
            float v = rsum[i];
            v += __shfl_xor(v, 1, 16);
            v += __shfl_xor(v, 2, 16);
            v += __shfl_xor(v, 4, 16);
            v += __shfl_xor(v, 8, 16);
            lrun[i] = lrun[i] * scal[i] + v;
        }
#pragma unroll
        for (int j = 0; j < 4; ++j)
#pragma unroll
            for (int i = 0; i < 8; ++i) oacc[j][i] *= scal[i];

        // ---- P (C-layout) -> LDS bf16 (row-major), then read as A-fragments ----
#pragma unroll
        for (int j = 0; j < 4; ++j)
#pragma unroll
            for (int i = 0; i < 8; ++i)
                Pb[(wave * 16 + rbase + i) * ALDS + j * 16 + l16] = f2bf(pv[j][i]);

        // Same-wave LDS write->read hazard: drain DS counter explicitly.
        asm volatile("s_wait_dscnt 0" ::: "memory");

        // ---- O += P @ V ----
#pragma unroll
        for (int ks = 0; ks < 2; ++ks) {
            FragB pa;
            const unsigned short* pp = Pb + (wave * 16 + l16) * ALDS + ks * 32 + koffA;
            pa.u[0] = *(const uint4*)(pp);
            pa.u[1] = *(const uint4*)(pp + 16);
#pragma unroll
            for (int jd = 0; jd < 4; ++jd) {
                FragB vb;
                const unsigned short* vp = Vt + (jd * 16 + l16) * ALDS + ks * 32 + koffA;
                vb.u[0] = *(const uint4*)(vp);
                vb.u[1] = *(const uint4*)(vp + 16);
                oacc[jd] = WMMA_BF16(pa.v, vb.v, oacc[jd]);
            }
        }
        __syncthreads();   // protect Kb/Vt before next iteration restage
    }

    // ---- Epilogue: renormalize (masked-softmax renorm w/ EPS) and store bf16 ----
#pragma unroll
    for (int jd = 0; jd < 4; ++jd) {
        int d = jd * 16 + l16;
#pragma unroll
        for (int i = 0; i < 8; ++i) {
            long t = rowQ + rbase + i;
            float v = oacc[jd][i] / (lrun[i] + 1e-13f);
            outb[t * 1024 + h * 64 + d] = f2bf(v);
        }
    }
}

// ---------------------------------------------------------------------------
// Host-side orchestration
// ---------------------------------------------------------------------------
extern "C" void kernel_launch(void* const* d_in, const int* in_sizes, int n_in,
                              void* d_out, int out_size, void* d_ws, size_t ws_size,
                              hipStream_t stream) {
    const int B = 8, T = 1024, D = 1024, O3 = 3072;
    const float* inputs = (const float*)d_in[0];   // (B,T,D)
    const float* mask   = (const float*)d_in[1];   // (B,T)
    const float* adj    = (const float*)d_in[2];   // (B,T,T)
    const float* W_qkv  = (const float*)d_in[3];   // (3072, D)
    const float* b_qkv  = (const float*)d_in[4];   // (3072,)
    const float* W_out  = (const float*)d_in[5];   // (D, 1024)
    const float* b_out  = (const float*)d_in[6];   // (D,)
    float* out = (float*)d_out;                    // (B,T,D) fp32

    char* ws = (char*)d_ws;
    unsigned short* Xbf   = (unsigned short*)ws; ws += (size_t)B * T * D * 2;     // 16 MB
    unsigned short* Wqkvb = (unsigned short*)ws; ws += (size_t)O3 * D * 2;        //  6 MB
    unsigned short* Woutb = (unsigned short*)ws; ws += (size_t)D * D * 2;         //  2 MB
    unsigned short* Projb = (unsigned short*)ws; ws += (size_t)B * T * O3 * 2;    // 48 MB
    unsigned short* Attb  = (unsigned short*)ws; ws += (size_t)B * T * D * 2;     // 16 MB

    // 1) fp32 -> bf16 conversions
    {
        int n4 = B * T * D / 4;
        cvt_f32_bf16<<<(n4 + 255) / 256, 256, 0, stream>>>(inputs, Xbf, n4);
        n4 = O3 * D / 4;
        cvt_f32_bf16<<<(n4 + 255) / 256, 256, 0, stream>>>(W_qkv, Wqkvb, n4);
        n4 = D * D / 4;
        cvt_f32_bf16<<<(n4 + 255) / 256, 256, 0, stream>>>(W_out, Woutb, n4);
    }

    // 2) QKV projection: (8192 x 1024) * (3072 x 1024)^T -> bf16 proj
    {
        dim3 grid(O3 / 128, (B * T) / 128);
        gemm_bf16nt<<<grid, 256, 0, stream>>>(Xbf, Wqkvb, b_qkv,
                                              nullptr, Projb, B * T, O3, D);
    }

    // 3) Fused attention: grid x = q-blocks (fast), z = batch (slow) so all 16
    //    heads of a batch run adjacently -> adj stays resident in 192MB L2.
    {
        dim3 grid(T_ / 64, 16, B);
        attn_fused<<<grid, 128, 0, stream>>>(Projb, adj, mask, Attb);
    }

    // 4) Output projection: (8192 x 1024) * (1024 x 1024)^T -> fp32 out
    {
        dim3 grid(D / 128, (B * T) / 128);
        gemm_bf16nt<<<grid, 256, 0, stream>>>(Attb, Woutb, b_out,
                                              out, nullptr, B * T, D, D);
    }
}
